// WindowAttention_51402168598846
// MI455X (gfx1250) — compile-verified, matrix operand order checked by reading, not run on hardware
//
#include <hip/hip_runtime.h>
#include <hip/hip_bf16.h>

typedef __attribute__((ext_vector_type(16))) _Float16 v16h;
typedef __attribute__((ext_vector_type(8)))  _Float16 v8h;
typedef __attribute__((ext_vector_type(8)))  float    v8f;
typedef __attribute__((ext_vector_type(2)))  float    f32x2;

#define NTOK   49
#define CDIM   192
#define NHEADS 6
#define HDIM   32
#define MPAD   64
#define NWIN   64
#define NBLK   4096

#define QKVW_ELEMS (3 * CDIM * CDIM)   // 110592
#define PROJW_ELEMS (CDIM * CDIM)      // 36864

// LDS strides (element units), chosen for 16B row alignment + bank spread
#define XS_STRIDE 200   // f16, x tile / O tile (aliased)
#define VT_STRIDE 72    // f16, v transposed [ch][tok]
#define S_STRIDE  68    // f32, attention scores
#define P_STRIDE  72    // f16, softmax probs

// LDS byte offsets
#define OFF_XS 0
#define OFF_Q  25600
#define OFF_K  51200
#define OFF_VT 76800
#define OFF_S  104448
#define OFF_P  121856
#define LDS_BYTES 131072

static __device__ __forceinline__ v8f wmma_f16(v16h a, v16h b, v8f c) {
    return __builtin_amdgcn_wmma_f32_16x16x32_f16(false, a, false, b,
                                                  (short)0, c, false, false);
}

// A/B operand fetch, 16-bit 16x32 fragment layout (ISA 7.12.2):
// lane: m|n = lane&15, half = lane>>4
// VGPR v: kbase = (v>>2)*16 + half*8 + (v&3)*2  -> two 16B loads
// Works for both LDS (ds_load_b128) and global f16 (global_load_b128) rows.
static __device__ __forceinline__ v16h load_frag(const _Float16* base, int row,
                                                 int stride, int koff, int half) {
    const _Float16* p = base + row * stride + koff + half * 8;
    v8h lo = *(const v8h*)(p);
    v8h hi = *(const v8h*)(p + 16);
    return __builtin_shufflevector(lo, hi, 0,1,2,3,4,5,6,7,8,9,10,11,12,13,14,15);
}

// B operand from a global f32 weight row (fallback path), f32->f16 on the fly
static __device__ __forceinline__ v16h load_b_f32(const float* wrow, int kstep, int half) {
    v16h r;
#pragma unroll
    for (int v = 0; v < 8; ++v) {
        int kb = kstep * 32 + ((v >> 2) << 4) + half * 8 + ((v & 3) << 1);
        f32x2 t = *(const f32x2*)(wrow + kb);
        r[2 * v]     = (_Float16)t.x;
        r[2 * v + 1] = (_Float16)t.y;
    }
    return r;
}

// One-shot weight conversion: f32 -> f16 into workspace (run once per launch)
__global__ __launch_bounds__(256)
void convert_weights(const float* __restrict__ qkv_w,
                     const float* __restrict__ proj_w,
                     _Float16* __restrict__ qkv16,
                     _Float16* __restrict__ proj16) {
    int i = blockIdx.x * 256 + threadIdx.x;
    if (i < QKVW_ELEMS)  qkv16[i]  = (_Float16)qkv_w[i];
    if (i < PROJW_ELEMS) proj16[i] = (_Float16)proj_w[i];
}

template <bool F16W>
__global__ __launch_bounds__(256)
void window_attn_fused(const float* __restrict__ x,
                       const float* __restrict__ mask,
                       const float* __restrict__ qkv_w,
                       const float* __restrict__ qkv_b,
                       const float* __restrict__ proj_w,
                       const float* __restrict__ proj_b,
                       const float* __restrict__ bias_table,
                       const int*   __restrict__ rel_index,
                       const _Float16* __restrict__ qkv_w16,
                       const _Float16* __restrict__ proj_w16,
                       float* __restrict__ out) {
    extern __shared__ char smem[];
    _Float16* xs  = (_Float16*)(smem + OFF_XS);   // x tile, later O tile
    _Float16* qs  = (_Float16*)(smem + OFF_Q);
    _Float16* ksm = (_Float16*)(smem + OFF_K);
    _Float16* vT  = (_Float16*)(smem + OFF_VT);
    float*    Sb  = (float*)(smem + OFF_S);
    _Float16* Pb  = (_Float16*)(smem + OFF_P);

    const int tid  = threadIdx.x;
    const int wave = tid >> 5;
    const int lane = tid & 31;
    const int ln   = lane & 15;
    const int half = lane >> 4;
    const int b    = blockIdx.x;
    const int w    = b & (NWIN - 1);
    const float* maskw = mask + (size_t)w * NTOK * NTOK;
    const float scale = 0.17677669529663687f;  // 32^-0.5

    // ---------------- Stage 0: load x window -> f16 LDS (zero-pad rows) -----
    const float* xb = x + (size_t)b * NTOK * CDIM;
    for (int i = tid; i < MPAD * CDIM; i += 256) {
        int r = i / CDIM, c = i - r * CDIM;
        float v = (r < NTOK) ? xb[r * CDIM + c] : 0.0f;
        xs[r * XS_STRIDE + c] = (_Float16)v;
    }
    __syncthreads();

    // ---------------- Stage 1: QKV GEMM (M=64, K=192, N=576) ----------------
    for (int nt = wave; nt < 36; nt += 8) {
        v8f acc[4];
#pragma unroll
        for (int mt = 0; mt < 4; ++mt) acc[mt] = (v8f){0,0,0,0,0,0,0,0};
        const int n_glob = nt * 16 + ln;
#pragma unroll
        for (int ksx = 0; ksx < 6; ++ksx) {
            v16h bm;
            if (F16W)
                bm = load_frag(qkv_w16, n_glob, CDIM, ksx * 32, half);
            else
                bm = load_b_f32(qkv_w + (size_t)n_glob * CDIM, ksx, half);
#pragma unroll
            for (int mt = 0; mt < 4; ++mt) {
                v16h am = load_frag(xs, mt * 16 + ln, XS_STRIDE, ksx * 32, half);
                acc[mt] = wmma_f16(am, bm, acc[mt]);
            }
        }
        const float qb = qkv_b[n_glob];
#pragma unroll
        for (int mt = 0; mt < 4; ++mt) {
#pragma unroll
            for (int r = 0; r < 8; ++r) {
                int m = mt * 16 + r + half * 8;
                float val = acc[mt][r] + qb;
                if (n_glob < 192) {
                    qs[m * XS_STRIDE + n_glob] = (_Float16)(val * scale);
                } else if (n_glob < 384) {
                    ksm[m * XS_STRIDE + (n_glob - 192)] = (_Float16)val;
                } else {
                    vT[(n_glob - 384) * VT_STRIDE + m] = (_Float16)val;
                }
            }
        }
    }
    __syncthreads();

    // ---------------- Stage 2: per-head attention ----------------------------
    for (int h = 0; h < NHEADS; ++h) {
        // ---- S = qs @ k^T + bias + mask   (16 tiles, 2 per wave) ----
        for (int t = wave * 2; t < wave * 2 + 2; ++t) {
            int mt  = t >> 2;
            int nt2 = t & 3;
            v16h aq = load_frag(qs,  mt  * 16 + ln, XS_STRIDE, h * HDIM, half);
            v16h bk = load_frag(ksm, nt2 * 16 + ln, XS_STRIDE, h * HDIM, half);
            v8f d = (v8f){0,0,0,0,0,0,0,0};
            d = wmma_f16(aq, bk, d);
            int n_g = nt2 * 16 + ln;
#pragma unroll
            for (int r = 0; r < 8; ++r) {
                int m_g = mt * 16 + r + half * 8;
                float val;
                if (n_g < NTOK) {
                    if (m_g < NTOK) {
                        int idx = rel_index[m_g * NTOK + n_g];
                        val = d[r] + bias_table[idx * NHEADS + h]
                                   + maskw[m_g * NTOK + n_g];
                    } else {
                        val = 0.0f;
                    }
                } else {
                    val = -1e30f;  // padded keys -> exp() == 0
                }
                Sb[m_g * S_STRIDE + n_g] = val;
            }
        }
        __syncthreads();

        // ---- row softmax -> Pb (f16, zero-padded) ----
        if (tid < MPAD) {
            int m = tid;
            if (m < NTOK) {
                float mx = -1e30f;
                for (int n = 0; n < NTOK; ++n)
                    mx = fmaxf(mx, Sb[m * S_STRIDE + n]);
                float sum = 0.0f;
                for (int n = 0; n < NTOK; ++n) {
                    float e = __expf(Sb[m * S_STRIDE + n] - mx);
                    Sb[m * S_STRIDE + n] = e;
                    sum += e;
                }
                float inv = 1.0f / sum;
                for (int n = 0; n < MPAD; ++n)
                    Pb[m * P_STRIDE + n] =
                        (n < NTOK) ? (_Float16)(Sb[m * S_STRIDE + n] * inv)
                                   : (_Float16)0.0f;
            } else {
                for (int n = 0; n < MPAD; ++n)
                    Pb[m * P_STRIDE + n] = (_Float16)0.0f;
            }
        }
        __syncthreads();

        // ---- O_h = P @ V   (8 tiles, 1 per wave; K=64 -> 2 wmma) ----
        {
            int mt  = wave >> 1;
            int nt2 = wave & 1;
            v8f d = (v8f){0,0,0,0,0,0,0,0};
#pragma unroll
            for (int ksx = 0; ksx < 2; ++ksx) {
                v16h ap = load_frag(Pb, mt * 16 + ln, P_STRIDE, ksx * 32, half);
                v16h bv = load_frag(vT, h * HDIM + nt2 * 16 + ln, VT_STRIDE,
                                    ksx * 32, half);
                d = wmma_f16(ap, bv, d);
            }
            int n_g = h * HDIM + nt2 * 16 + ln;  // output channel
#pragma unroll
            for (int r = 0; r < 8; ++r) {
                int m_g = mt * 16 + r + half * 8;
                xs[m_g * XS_STRIDE + n_g] = (_Float16)d[r];  // O aliases xs
            }
        }
        __syncthreads();
    }

    // ---------------- Stage 3: out = O @ proj_w^T + proj_b ------------------
    float* outb = out + (size_t)b * NTOK * CDIM;
    for (int nt = wave; nt < 12; nt += 8) {
        v8f acc[4];
#pragma unroll
        for (int mt = 0; mt < 4; ++mt) acc[mt] = (v8f){0,0,0,0,0,0,0,0};
        const int n_glob = nt * 16 + ln;
#pragma unroll
        for (int ksx = 0; ksx < 6; ++ksx) {
            v16h bm;
            if (F16W)
                bm = load_frag(proj_w16, n_glob, CDIM, ksx * 32, half);
            else
                bm = load_b_f32(proj_w + (size_t)n_glob * CDIM, ksx, half);
#pragma unroll
            for (int mt = 0; mt < 4; ++mt) {
                v16h am = load_frag(xs, mt * 16 + ln, XS_STRIDE, ksx * 32, half);
                acc[mt] = wmma_f16(am, bm, acc[mt]);
            }
        }
        const float pb = proj_b[n_glob];
#pragma unroll
        for (int mt = 0; mt < 4; ++mt) {
#pragma unroll
            for (int r = 0; r < 8; ++r) {
                int m_g = mt * 16 + r + half * 8;
                if (m_g < NTOK)
                    outb[m_g * CDIM + n_glob] = acc[mt][r] + pb;
            }
        }
    }
}

extern "C" void kernel_launch(void* const* d_in, const int* in_sizes, int n_in,
                              void* d_out, int out_size, void* d_ws, size_t ws_size,
                              hipStream_t stream) {
    const float* x          = (const float*)d_in[0];
    const float* mask       = (const float*)d_in[1];
    const float* qkv_w      = (const float*)d_in[2];
    const float* qkv_b      = (const float*)d_in[3];
    const float* proj_w     = (const float*)d_in[4];
    const float* proj_b     = (const float*)d_in[5];
    const float* bias_table = (const float*)d_in[6];
    const int*   rel_index  = (const int*)d_in[7];
    float* out = (float*)d_out;

    dim3 grid(NBLK), block(256);
    const size_t need = (size_t)(QKVW_ELEMS + PROJW_ELEMS) * sizeof(_Float16);

    if (ws_size >= need) {
        _Float16* qkv16  = (_Float16*)d_ws;
        _Float16* proj16 = qkv16 + QKVW_ELEMS;
        convert_weights<<<(QKVW_ELEMS + 255) / 256, 256, 0, stream>>>(
            qkv_w, proj_w, qkv16, proj16);
        window_attn_fused<true><<<grid, block, LDS_BYTES, stream>>>(
            x, mask, qkv_w, qkv_b, proj_w, proj_b, bias_table, rel_index,
            qkv16, proj16, out);
    } else {
        window_attn_fused<false><<<grid, block, LDS_BYTES, stream>>>(
            x, mask, qkv_w, qkv_b, proj_w, proj_b, bias_table, rel_index,
            (const _Float16*)nullptr, (const _Float16*)nullptr, out);
    }
}
